// CausalSelfAttention_50139448213548
// MI455X (gfx1250) — compile-verified, matrix operand order checked by reading, not run on hardware
//
#include <hip/hip_runtime.h>
#include <hip/hip_bf16.h>

typedef __attribute__((ext_vector_type(16))) __bf16 v16bf;
typedef __attribute__((ext_vector_type(8)))  float  v8f;
typedef __attribute__((ext_vector_type(4)))  int    v4i;

typedef __attribute__((address_space(1))) v4i* gas_v4i;
typedef __attribute__((address_space(3))) v4i* las_v4i;

#define C_EMB 1024
#define T_SEQ 2048
#define N_B   4
#define N_H   16
#define H_D   64
#define N3    3072
#define M_TOT (N_B * T_SEQ)   // 8192
#define WPAD  48              // padded LDS row stride (96B) to break bank conflicts

union FragBF {
    v16bf v;
    uint4 u[2];
};

__device__ __forceinline__ v8f wmma_bf16(const FragBF& a, const FragBF& b, v8f c) {
    return __builtin_amdgcn_wmma_f32_16x16x32_bf16(false, a.v, false, b.v, (short)0, c, false, false);
}

// gfx1250 async global->LDS copy (ASYNCcnt-tracked), 16 bytes per lane
__device__ __forceinline__ void async_b128(const __bf16* gsrc, __bf16* ldst) {
    __builtin_amdgcn_global_load_async_to_lds_b128(
        (gas_v4i)gsrc, (las_v4i)ldst, 0, 0);
}

__device__ __forceinline__ FragBF load_a_frag(const __bf16* row, int kb, int half) {
    FragBF f;
    f.u[0] = *reinterpret_cast<const uint4*>(row + kb + half * 8);
    f.u[1] = *reinterpret_cast<const uint4*>(row + kb + 16 + half * 8);
    return f;
}

// ---------------- Stage 1: precision conversion / weight transpose ----------------

__global__ void cvt_f32_bf16(const float* __restrict__ src, __bf16* __restrict__ dst, int n4) {
    int i = blockIdx.x * blockDim.x + threadIdx.x;
    if (i >= n4) return;
    float4 f = reinterpret_cast<const float4*>(src)[i];
    union { __bf16 h[4]; uint2 u; } o;
    o.h[0] = (__bf16)f.x; o.h[1] = (__bf16)f.y; o.h[2] = (__bf16)f.z; o.h[3] = (__bf16)f.w;
    reinterpret_cast<uint2*>(dst)[i] = o.u;
}

// dst[n*1024 + k] = (bf16) src[k*cols + n]   (K dim is always 1024 here)
__global__ void transpose_w(const float* __restrict__ src, __bf16* __restrict__ dst,
                            int cols, int total) {
    int i = blockIdx.x * blockDim.x + threadIdx.x;
    if (i >= total) return;
    int n = i >> 10;
    int k = i & 1023;
    dst[i] = (__bf16)src[(size_t)k * cols + n];
}

// ---------------- shared GEMM core: 128(M) x 64(N) per block, B staged in LDS -----
// Computes acc[4] (nt = 0..3 tiles of 16 cols) for this wave's 16-row strip.
// Weight tile (64 x 32 bf16) double-buffered in LDS via async global->LDS copies.

__device__ __forceinline__ void gemm_core(
    const __bf16* __restrict__ A, const __bf16* __restrict__ Wt,
    __bf16 (*wtile)[64 * WPAD],   // [2][64*WPAD] LDS
    int m0, int n0, int r, int half, v8f acc[4])
{
    const int tid  = threadIdx.x;
    const int crow = tid >> 2;   // 0..63 : weight row within tile
    const int cchk = tid & 3;    // 0..3  : 16B chunk within 64B row
    const __bf16* wsrc = Wt + (size_t)(n0 + crow) * 1024 + cchk * 8;
    __bf16* wdst0 = &wtile[0][crow * WPAD + cchk * 8];
    __bf16* wdst1 = &wtile[1][crow * WPAD + cchk * 8];
    const __bf16* arow = A + (size_t)(m0 + r) * 1024;

    async_b128(wsrc, wdst0);                 // tile for kb = 0
    FragBF a = load_a_frag(arow, 0, half);   // A frag for kb = 0

    for (int kb = 0; kb < 1024 - 32; kb += 32) {
        const int cur = (kb >> 5) & 1;
        async_b128(wsrc + kb + 32, cur ? wdst0 : wdst1);   // prefetch next tile
        __builtin_amdgcn_s_wait_asynccnt(1);               // current tile landed
        __syncthreads();                                   // ... in every wave

        FragBF an = load_a_frag(arow, kb + 32, half);      // prefetch next A frag

        const __bf16* lb = wtile[cur];
#pragma unroll
        for (int nt = 0; nt < 4; ++nt) {
            FragBF bf;
            const __bf16* lrow = lb + (nt * 16 + r) * WPAD;
            bf.u[0] = *reinterpret_cast<const uint4*>(lrow + half * 8);
            bf.u[1] = *reinterpret_cast<const uint4*>(lrow + 16 + half * 8);
            acc[nt] = wmma_bf16(a, bf, acc[nt]);
        }
        a = an;
        __syncthreads();   // everyone done reading before buffer is overwritten
    }
    // peeled last K-tile (kb = 992, buffer index 1)
    __builtin_amdgcn_s_wait_asynccnt(0);
    __syncthreads();
    {
        const __bf16* lb = wtile[1];
#pragma unroll
        for (int nt = 0; nt < 4; ++nt) {
            FragBF bf;
            const __bf16* lrow = lb + (nt * 16 + r) * WPAD;
            bf.u[0] = *reinterpret_cast<const uint4*>(lrow + half * 8);
            bf.u[1] = *reinterpret_cast<const uint4*>(lrow + 16 + half * 8);
            acc[nt] = wmma_bf16(a, bf, acc[nt]);
        }
    }
}

// ---------------- Stage 2: QKV GEMM  [8192,1024] x [1024,3072] -> Q/K/Vt bf16 -----

__global__ __launch_bounds__(256, 1) void qkv_gemm(
    const __bf16* __restrict__ A,      // x, bf16 [8192,1024]
    const __bf16* __restrict__ Wt,     // qkv_w^T bf16 [3072,1024]
    const float*  __restrict__ bias,   // [3072]
    __bf16* __restrict__ Q,            // [B,H,T,D]  (pre-scaled by 0.125)
    __bf16* __restrict__ K,            // [B,H,T,D]
    __bf16* __restrict__ Vt)           // [B,H,D,T]
{
    __shared__ __align__(16) __bf16 wtile[2][64 * WPAD];
    const int wave = threadIdx.x >> 5;
    const int lane = threadIdx.x & 31;
    const int r    = lane & 15;
    const int half = lane >> 4;
    const int n0   = blockIdx.x * 64;
    const int m0   = (blockIdx.y * 8 + wave) * 16;

    v8f acc[4] = {};
    gemm_core(A, Wt, wtile, m0, n0, r, half, acc);

    const int s = n0 >> 10;            // 0=Q 1=K 2=V (uniform per block)
    const int h = (n0 >> 6) & 15;      // head (uniform per block)
#pragma unroll
    for (int nt = 0; nt < 4; ++nt) {
        const int col = n0 + nt * 16 + r;
        const int d   = col & 63;
        const float bv = bias[col];
#pragma unroll
        for (int i = 0; i < 8; ++i) {
            const int m = m0 + i + 8 * half;
            const int b = m >> 11;
            const int t = m & 2047;
            const int bh = b * 16 + h;
            float v = acc[nt][i] + bv;
            if (s == 0)
                Q[((size_t)bh * 2048 + t) * 64 + d] = (__bf16)(v * 0.125f);
            else if (s == 1)
                K[((size_t)bh * 2048 + t) * 64 + d] = (__bf16)v;
            else
                Vt[((size_t)bh * 64 + d) * 2048 + t] = (__bf16)v;
        }
    }
}

// ---------------- Stage 3: fused causal flash attention --------------------------

__global__ __launch_bounds__(256, 1) void attn_flash(
    const __bf16* __restrict__ Q,   // [B,H,T,D] pre-scaled
    const __bf16* __restrict__ K,   // [B,H,T,D]
    const __bf16* __restrict__ Vt,  // [B,H,D,T]
    __bf16* __restrict__ O)         // [B,T,C]
{
    __shared__ __align__(16) __bf16 ldsP[8][16 * WPAD];

    const int wave = threadIdx.x >> 5;
    const int lane = threadIdx.x & 31;
    const int r    = lane & 15;
    const int half = lane >> 4;
    const int gid  = blockIdx.x * 8 + wave;
    const int bh   = gid >> 7;          // (T/16)=128 q-tiles per head
    const int q0   = (gid & 127) * 16;

    const __bf16* Qb = Q  + (size_t)bh * 2048 * 64;
    const __bf16* Kb = K  + (size_t)bh * 2048 * 64;
    const __bf16* Vb = Vt + (size_t)bh * 64 * 2048;

    // Q strip: 2 A-fragments covering d = 0..63, loaded once
    FragBF qf[2];
    {
        const __bf16* qrow = Qb + (size_t)(q0 + r) * 64;
#pragma unroll
        for (int ks = 0; ks < 2; ++ks) {
            qf[ks].u[0] = *reinterpret_cast<const uint4*>(qrow + ks * 32 + half * 8);
            qf[ks].u[1] = *reinterpret_cast<const uint4*>(qrow + ks * 32 + 16 + half * 8);
        }
    }

    v8f oacc[4] = {};
    float rm[8], rl[8];
#pragma unroll
    for (int i = 0; i < 8; ++i) { rm[i] = -1e30f; rl[i] = 0.0f; }

    for (int jb = 0; jb <= q0 + 15; jb += 32) {
        // --- issue K fragment loads first (needed soonest) ---
        FragBF kf[2][2];
#pragma unroll
        for (int st = 0; st < 2; ++st) {
#pragma unroll
            for (int ks = 0; ks < 2; ++ks) {
                const __bf16* krow = Kb + (size_t)(jb + st * 16 + r) * 64 + ks * 32;
                kf[st][ks].u[0] = *reinterpret_cast<const uint4*>(krow + half * 8);
                kf[st][ks].u[1] = *reinterpret_cast<const uint4*>(krow + 16 + half * 8);
            }
        }
        // --- then V fragments: their latency hides behind softmax ---
        FragBF vf[4];
#pragma unroll
        for (int dt = 0; dt < 4; ++dt) {
            const __bf16* vrow = Vb + (size_t)(dt * 16 + r) * 2048 + jb;
            vf[dt].u[0] = *reinterpret_cast<const uint4*>(vrow + half * 8);
            vf[dt].u[1] = *reinterpret_cast<const uint4*>(vrow + 16 + half * 8);
        }
        // prefetch next KV block's K rows into cache
        if (jb + 32 <= q0 + 15) {
            __builtin_prefetch(Kb + (size_t)(jb + 32 + r) * 64, 0, 0);
            __builtin_prefetch(Kb + (size_t)(jb + 48 + r) * 64, 0, 0);
        }

        // S = Q * K^T for 32 kv columns (two 16x16 tiles)
        v8f sac[2] = {};
#pragma unroll
        for (int st = 0; st < 2; ++st)
#pragma unroll
            for (int ks = 0; ks < 2; ++ks)
                sac[st] = wmma_bf16(qf[ks], kf[st][ks], sac[st]);

        // causal mask + online softmax; each row lives in 16 lanes of one half
#pragma unroll
        for (int i = 0; i < 8; ++i) {
            const int t  = q0 + i + 8 * half;
            const int j0 = jb + r;
            float s0 = (j0      <= t) ? sac[0][i] : -__builtin_inff();
            float s1 = (j0 + 16 <= t) ? sac[1][i] : -__builtin_inff();

            float mx = fmaxf(s0, s1);
#pragma unroll
            for (int off = 1; off < 16; off <<= 1)
                mx = fmaxf(mx, __shfl_xor(mx, off, 16));

            const float nm   = fmaxf(rm[i], mx);
            const float corr = __expf(rm[i] - nm);
            const float p0   = __expf(s0 - nm);
            const float p1   = __expf(s1 - nm);
            float ps = p0 + p1;
#pragma unroll
            for (int off = 1; off < 16; off <<= 1)
                ps += __shfl_xor(ps, off, 16);

            rl[i] = rl[i] * corr + ps;
            rm[i] = nm;
#pragma unroll
            for (int dt = 0; dt < 4; ++dt) oacc[dt][i] *= corr;

            // stash P tile (C-frag layout -> LDS row-major, padded stride)
            ldsP[wave][(i + 8 * half) * WPAD + r]      = (__bf16)p0;
            ldsP[wave][(i + 8 * half) * WPAD + 16 + r] = (__bf16)p1;
        }
        __builtin_amdgcn_wave_barrier();
        asm volatile("" ::: "memory");   // same-wave LDS ops are in-order

        // reread P as an A-fragment
        FragBF pf;
        pf.u[0] = *reinterpret_cast<const uint4*>(&ldsP[wave][r * WPAD + half * 8]);
        pf.u[1] = *reinterpret_cast<const uint4*>(&ldsP[wave][r * WPAD + 16 + half * 8]);

        // O += P * V (V fragments already resident)
#pragma unroll
        for (int dt = 0; dt < 4; ++dt)
            oacc[dt] = wmma_bf16(pf, vf[dt], oacc[dt]);
    }

    // normalize and write [B,T,C] bf16
    const int b = bh >> 4, h = bh & 15;
#pragma unroll
    for (int i = 0; i < 8; ++i) {
        const float inv = 1.0f / rl[i];
        const int t = q0 + i + 8 * half;
        const size_t row = ((size_t)b * 2048 + t) * 1024 + h * 64;
#pragma unroll
        for (int dt = 0; dt < 4; ++dt)
            O[row + dt * 16 + r] = (__bf16)(oacc[dt][i] * inv);
    }
}

// ---------------- Stage 4: output projection  [8192,1024] x [1024,1024] ----------

__global__ __launch_bounds__(256, 1) void proj_gemm(
    const __bf16* __restrict__ A,     // attn out bf16 [8192,1024]
    const __bf16* __restrict__ Wt,    // out_w^T bf16 [1024,1024]
    const float*  __restrict__ bias,  // [1024]
    float* __restrict__ out)          // fp32 [8192,1024]
{
    __shared__ __align__(16) __bf16 wtile[2][64 * WPAD];
    const int wave = threadIdx.x >> 5;
    const int lane = threadIdx.x & 31;
    const int r    = lane & 15;
    const int half = lane >> 4;
    const int n0   = blockIdx.x * 64;
    const int m0   = (blockIdx.y * 8 + wave) * 16;

    v8f acc[4] = {};
    gemm_core(A, Wt, wtile, m0, n0, r, half, acc);

#pragma unroll
    for (int nt = 0; nt < 4; ++nt) {
        const int col = n0 + nt * 16 + r;
        const float bv = bias[col];
#pragma unroll
        for (int i = 0; i < 8; ++i) {
            const int m = m0 + i + 8 * half;
            out[(size_t)m * 1024 + col] = acc[nt][i] + bv;
        }
    }
}

// ---------------- launch ----------------------------------------------------------

extern "C" void kernel_launch(void* const* d_in, const int* in_sizes, int n_in,
                              void* d_out, int out_size, void* d_ws, size_t ws_size,
                              hipStream_t stream) {
    const float* x      = (const float*)d_in[0];
    const float* qkv_w  = (const float*)d_in[1];
    const float* qkv_b  = (const float*)d_in[2];
    const float* out_w  = (const float*)d_in[3];
    const float* out_b  = (const float*)d_in[4];
    float* out = (float*)d_out;

    size_t off = 0;
    auto alloc = [&](size_t bytes) -> void* {
        void* p = (char*)d_ws + off;
        off += (bytes + 255) & ~(size_t)255;
        return p;
    };
    __bf16* xb    = (__bf16*)alloc((size_t)M_TOT * C_EMB * 2);          // 16 MB
    __bf16* wqkvT = (__bf16*)alloc((size_t)N3 * C_EMB * 2);             // 6 MB
    __bf16* woutT = (__bf16*)alloc((size_t)C_EMB * C_EMB * 2);          // 2 MB
    __bf16* Qb    = (__bf16*)alloc((size_t)N_B * N_H * T_SEQ * H_D * 2);// 16 MB
    __bf16* Kb    = (__bf16*)alloc((size_t)N_B * N_H * T_SEQ * H_D * 2);
    __bf16* Vtb   = (__bf16*)alloc((size_t)N_B * N_H * H_D * T_SEQ * 2);
    __bf16* Ab    = (__bf16*)alloc((size_t)M_TOT * C_EMB * 2);          // 16 MB

    // Stage 1
    {
        int n4 = M_TOT * C_EMB / 4;
        cvt_f32_bf16<<<n4 / 256, 256, 0, stream>>>(x, xb, n4);
        int t1 = N3 * C_EMB;
        transpose_w<<<t1 / 256, 256, 0, stream>>>(qkv_w, wqkvT, N3, t1);
        int t2 = C_EMB * C_EMB;
        transpose_w<<<t2 / 256, 256, 0, stream>>>(out_w, woutT, C_EMB, t2);
    }
    // Stage 2: QKV projection -> Q / K / V^T (bf16, head-major)
    qkv_gemm<<<dim3(N3 / 64, M_TOT / 128), 256, 0, stream>>>(xb, wqkvT, qkv_b, Qb, Kb, Vtb);
    // Stage 3: fused causal attention
    attn_flash<<<(N_B * N_H * (T_SEQ / 16)) / 8, 256, 0, stream>>>(Qb, Kb, Vtb, Ab);
    // Stage 4: output projection (fp32 out + bias)
    proj_gemm<<<dim3(C_EMB / 64, M_TOT / 128), 256, 0, stream>>>(Ab, woutT, out_b, out);
}
